// GAT3_7730941133134
// MI455X (gfx1250) — matrix operand order repaired and checked by reading, not compiled, not run on hardware
//
#include <hip/hip_runtime.h>
#include <math.h>

// MI455X (gfx1250): wave32, WMMA 16x16x32 f16 -> f32 accumulate.
typedef __attribute__((ext_vector_type(16))) _Float16 v16h;
typedef __attribute__((ext_vector_type(8)))  float    v8f;

// ---------------- device helpers ----------------

__device__ __forceinline__ void atomicMaxF(float* addr, float val) {
  // ordered-float trick: positives compare as ints, negatives as reversed uints
  if (val >= 0.0f) atomicMax((int*)addr, __float_as_int(val));
  else             atomicMin((unsigned int*)addr, (unsigned int)__float_as_int(val));
}

__device__ __forceinline__ void edge_sd(const int* __restrict__ ei, int E, int e,
                                        int& s, int& d) {
  if (e < E) { s = ei[e]; d = ei[E + e]; }     // edge_index rows: [src | dst]
  else       { s = e - E; d = e - E; }         // appended self-loops
}

// ---------------- WMMA GEMM: C[M,N] = A[M,K] @ B[K,N] ----------------
// K, N compile-time: branch-free, immediate-offset addressing, k-loop unrolled.
// One 16x16 C tile per wave; 8 waves / 256-thread block.
// A tile staged through wave-private LDS (coalesced b128 global loads -> f16).

template <int K, int N>
__global__ __launch_bounds__(256) void gemm_wmma(
    const float* __restrict__ A, const float* __restrict__ B,
    float* __restrict__ C, int M) {
  __shared__ _Float16 lA[8][16 * 32];                // 1 KB per wave

  const int lane = threadIdx.x & 31;
  const int wave = threadIdx.x >> 5;
  constexpr int tilesN = N / 16;
  const int tilesM = (M + 15) >> 4;

  int tile = blockIdx.x * 8 + wave;
  const int lastTile = tilesM * tilesN - 1;
  if (tile > lastTile) tile = lastTile;              // clamp: redundant identical work,
                                                     // keeps EXEC all-1s for WMMA
  const int tm = tile / tilesN;
  const int tn = tile - tm * tilesN;
  const int mBase = tm << 4;
  const int nBase = tn << 4;
  const int half = lane >> 4;                        // lanes 0-15 / 16-31
  const int lid  = lane & 15;
  const int nCol = nBase + lid;

  // A staging: lane loads row (lane>>1), 16 contiguous floats at col (lane&1)*16
  const int ar  = lane >> 1;
  const int ac0 = (lane & 1) << 4;
  const float* __restrict__ aRow = A + (size_t)(mBase + ar) * K + ac0;
  _Float16* __restrict__ lAw = lA[wave];

  // B: element e of k-step k0 lives at row (k0 + half*16 + e), col nCol
  const float* __restrict__ bp = B + (size_t)(half << 4) * N + nCol;

  v8f acc = {};
#pragma unroll
  for (int k0 = 0; k0 < K; k0 += 32) {
    // ---- stage A tile (16x32 f32 -> f16 in LDS), fully coalesced ----
    const float4* __restrict__ aVec = (const float4*)(aRow + k0);
#pragma unroll
    for (int j = 0; j < 4; ++j) {
      float4 v = aVec[j];
      const int base = ar * 32 + ac0 + j * 4;
      lAw[base + 0] = (_Float16)v.x;
      lAw[base + 1] = (_Float16)v.y;
      lAw[base + 2] = (_Float16)v.z;
      lAw[base + 3] = (_Float16)v.w;
    }
    __builtin_amdgcn_wave_barrier();                 // wave-private LDS: order only

    v16h av, bv;
#pragma unroll
    for (int e = 0; e < 16; ++e) {
      // 16-bit A 16x32 layout (contiguous, matches B): K = half*16 + e
      const int ka = (half << 4) + e;
      av[e] = lAw[lid * 32 + ka];
      // B k order used here (contiguous, matches A): lanes 0-15 K=0..15, lanes 16-31 K=16..31
      bv[e] = (_Float16)bp[(size_t)(k0 + e) * N];    // immediate offsets (N const)
    }
    __builtin_amdgcn_wave_barrier();
    acc = __builtin_amdgcn_wmma_f32_16x16x32_f16(false, av, false, bv,
                                                 (short)0, acc, false, false);
  }

  // C 16x16 f32 layout: lane = N col; VGPR r -> M = half*8 + r
  const int mOut = mBase + (half << 3);
#pragma unroll
  for (int r = 0; r < 8; ++r)
    C[(size_t)(mOut + r) * N + nCol] = acc[r];
}

// ---------------- small kernels ----------------

__global__ void fill_f32(float* __restrict__ p, float v, long long n) {
  long long i = (long long)blockIdx.x * blockDim.x + threadIdx.x;
  if (i < n) p[i] = v;
}

// pad W[K,Nin] -> WP[K,Npad] (zero fill)
__global__ void pad_cols(const float* __restrict__ W, float* __restrict__ WP,
                         int K, int Nin, int Npad) {
  int idx = blockIdx.x * blockDim.x + threadIdx.x;
  if (idx >= K * Npad) return;
  int k = idx / Npad, j = idx - k * Npad;
  WP[idx] = (j < Nin) ? W[k * Nin + j] : 0.0f;
}

// compact src[n,ldsrc] -> dst[n,cols]
__global__ void compact_cols(const float* __restrict__ src, float* __restrict__ dst,
                             int n, int ldsrc, int cols) {
  int idx = blockIdx.x * blockDim.x + threadIdx.x;
  if (idx >= n * cols) return;
  int r = idx / cols, c = idx - r * cols;
  dst[idx] = src[(size_t)r * ldsrc + c];
}

// per-node attention logits: al[n,h] = sum_c H[n,h,c] * a[h,c]
__global__ void node_logits(const float* __restrict__ H, const float* __restrict__ aS,
                            const float* __restrict__ aD, float* __restrict__ alS,
                            float* __restrict__ alD, int n, int heads, int oc) {
  int idx = blockIdx.x * blockDim.x + threadIdx.x;
  if (idx >= n * heads) return;
  int node = idx / heads, h = idx - node * heads;
  const float* hp = H + (size_t)node * heads * oc + (size_t)h * oc;
  const float* as = aS + h * oc;
  const float* ad = aD + h * oc;
  float ss = 0.f, sd = 0.f;
  for (int c = 0; c < oc; ++c) { float v = hp[c]; ss += v * as[c]; sd += v * ad[c]; }
  alS[idx] = ss; alD[idx] = sd;
}

// phase 1: e = leaky_relu(al_s[src]+al_d[dst]); store score; segment max into Mb
__global__ void edge_score(const int* __restrict__ ei, int E, int ET,
                           const float* __restrict__ alS, const float* __restrict__ alD,
                           float* __restrict__ P, float* __restrict__ Mb, int heads) {
  long long idx = (long long)blockIdx.x * blockDim.x + threadIdx.x;
  long long tot = (long long)ET * heads;
  if (idx >= tot) return;
  int e = (int)(idx / heads), h = (int)(idx - (long long)e * heads);
  int s, d; edge_sd(ei, E, e, s, d);
  float v = alS[s * heads + h] + alD[d * heads + h];
  v = (v > 0.f) ? v : 0.2f * v;                      // leaky_relu, slope 0.2
  P[idx] = v;
  atomicMaxF(&Mb[d * heads + h], v);
}

// phase 2: p = exp(e - m[dst]); segment sum into Sb; overwrite P with p
__global__ void edge_softmax_p(const int* __restrict__ ei, int E, int ET,
                               float* __restrict__ P, const float* __restrict__ Mb,
                               float* __restrict__ Sb, int heads) {
  long long idx = (long long)blockIdx.x * blockDim.x + threadIdx.x;
  long long tot = (long long)ET * heads;
  if (idx >= tot) return;
  int e = (int)(idx / heads), h = (int)(idx - (long long)e * heads);
  int s, d; edge_sd(ei, E, e, s, d);
  (void)s;
  float p = expf(P[idx] - Mb[d * heads + h]);
  P[idx] = p;
  atomicAdd(&Sb[d * heads + h], p);
}

// phase 3: out[dst, c] += (p / (s[dst]+eps)) * H[src, c]   (c spans heads*oc)
__global__ void edge_agg(const int* __restrict__ ei, int E, int ET,
                         const float* __restrict__ H, const float* __restrict__ P,
                         const float* __restrict__ Sb, float* __restrict__ out,
                         int heads, int oc) {
  long long idx = (long long)blockIdx.x * blockDim.x + threadIdx.x;
  const int CT = heads * oc;
  long long tot = (long long)ET * CT;
  if (idx >= tot) return;
  int e = (int)(idx / CT), c = (int)(idx - (long long)e * CT);
  int h = c / oc;
  int s, d; edge_sd(ei, E, e, s, d);
  float alpha = P[(long long)e * heads + h] / (Sb[d * heads + h] + 1e-16f);
  atomicAdd(&out[(size_t)d * CT + c], alpha * H[(size_t)s * CT + c]);
}

// epilogue: y = act(agg + bias); ELU for layers 1-2
__global__ void bias_act(const float* __restrict__ in, const float* __restrict__ b,
                         float* __restrict__ out, long long total, int CT, int do_elu) {
  long long idx = (long long)blockIdx.x * blockDim.x + threadIdx.x;
  if (idx >= total) return;
  float v = in[idx] + b[idx % CT];
  if (do_elu) v = (v > 0.f) ? v : (expf(v) - 1.f);
  out[idx] = v;
}

// ---------------- host-side drivers ----------------

static inline unsigned blks(long long total, int t) {
  return (unsigned)((total + t - 1) / t);
}

// post-GEMM stages: logits -> 3-phase edge softmax -> aggregate -> bias(+ELU)
static void run_attention(hipStream_t stream, const int* ei, int n, int E, int ET,
                          const float* H, const float* aS, const float* aD,
                          const float* bias, int heads, int oc, int do_elu,
                          float* ALS, float* ALD, float* Mb, float* Sb, float* P,
                          float* AGG, float* Y) {
  const int CT = heads * oc;
  node_logits<<<blks((long long)n * heads, 256), 256, 0, stream>>>(H, aS, aD, ALS, ALD, n, heads, oc);
  fill_f32<<<blks((long long)n * heads, 256), 256, 0, stream>>>(Mb, -__builtin_inff(), (long long)n * heads);
  fill_f32<<<blks((long long)n * heads, 256), 256, 0, stream>>>(Sb, 0.0f, (long long)n * heads);
  fill_f32<<<blks((long long)n * CT, 256), 256, 0, stream>>>(AGG, 0.0f, (long long)n * CT);
  const long long eh = (long long)ET * heads;
  edge_score<<<blks(eh, 256), 256, 0, stream>>>(ei, E, ET, ALS, ALD, P, Mb, heads);
  edge_softmax_p<<<blks(eh, 256), 256, 0, stream>>>(ei, E, ET, P, Mb, Sb, heads);
  const long long ec = (long long)ET * CT;
  edge_agg<<<blks(ec, 256), 256, 0, stream>>>(ei, E, ET, H, P, Sb, AGG, heads, oc);
  bias_act<<<blks((long long)n * CT, 256), 256, 0, stream>>>(AGG, bias, Y, (long long)n * CT, CT, do_elu);
}

extern "C" void kernel_launch(void* const* d_in, const int* in_sizes, int n_in,
                              void* d_out, int out_size, void* d_ws, size_t ws_size,
                              hipStream_t stream) {
  const float* x   = (const float*)d_in[0];
  const int*   ei  = (const int*)  d_in[1];
  const float* W1  = (const float*)d_in[2];
  const float* aS1 = (const float*)d_in[3];
  const float* aD1 = (const float*)d_in[4];
  const float* b1  = (const float*)d_in[5];
  const float* W2  = (const float*)d_in[6];
  const float* aS2 = (const float*)d_in[7];
  const float* aD2 = (const float*)d_in[8];
  const float* b2  = (const float*)d_in[9];
  const float* W3  = (const float*)d_in[10];
  const float* aS3 = (const float*)d_in[11];
  const float* aD3 = (const float*)d_in[12];
  const float* b3  = (const float*)d_in[13];

  const int n  = in_sizes[0] / 32;     // 10000 (multiple of 16)
  const int E  = in_sizes[1] / 2;      // 160000
  const int ET = E + n;                // + self-loops

  // scratch carve (~49 MB total -> L2 resident)
  char* ws = (char*)d_ws;
  size_t off = 0;
  auto carve = [&](size_t bytes) -> char* {
    char* p = ws + off;
    off = (off + bytes + 255) & ~(size_t)255;
    return p;
  };
  float* H    = (float*)carve((size_t)n * 512 * 4);   // projection output
  float* ACT  = (float*)carve((size_t)n * 512 * 4);   // activations / agg output
  float* ALS  = (float*)carve((size_t)n * 8 * 4);
  float* ALD  = (float*)carve((size_t)n * 8 * 4);
  float* Mb   = (float*)carve((size_t)n * 8 * 4);
  float* Sb   = (float*)carve((size_t)n * 8 * 4);
  float* P    = (float*)carve((size_t)ET * 8 * 4);
  float* HPAD = (float*)carve((size_t)n * 16 * 4);    // layer-3 padded projection
  float* OUT3 = (float*)carve((size_t)n * 16 * 4);    // layer-3 aggregate
  float* WPAD = (float*)carve((size_t)512 * 16 * 4);  // layer-3 padded weights
  (void)ws_size; (void)n_in; (void)out_size;

  const int tilesM = (n + 15) / 16;

  // ---- layer 1: 32 -> 8 heads x 8, concat, ELU ----
  gemm_wmma<32, 64><<<blks(tilesM * 4, 8), 256, 0, stream>>>(x, W1, H, n);
  run_attention(stream, ei, n, E, ET, H, aS1, aD1, b1, 8, 8, 1,
                ALS, ALD, Mb, Sb, P, ACT, ACT);

  // ---- layer 2: 64 -> 8 heads x 64, concat, ELU ----
  gemm_wmma<64, 512><<<blks(tilesM * 32, 8), 256, 0, stream>>>(ACT, W2, H, n);
  run_attention(stream, ei, n, E, ET, H, aS2, aD2, b2, 8, 64, 1,
                ALS, ALD, Mb, Sb, P, ACT, ACT);

  // ---- layer 3: 512 -> 1 head x 6, mean(1 head)=identity, no activation ----
  pad_cols<<<blks(512 * 16, 256), 256, 0, stream>>>(W3, WPAD, 512, 6, 16);
  gemm_wmma<512, 16><<<blks(tilesM * 1, 8), 256, 0, stream>>>(ACT, WPAD, HPAD, n);
  compact_cols<<<blks((long long)n * 6, 256), 256, 0, stream>>>(HPAD, H, n, 16, 6);
  run_attention(stream, ei, n, E, ET, H, aS3, aD3, b3, 1, 6, 0,
                ALS, ALD, Mb, Sb, P, OUT3, (float*)d_out);
}
